// CorticalMap_26396869001649
// MI455X (gfx1250) — compile-verified, matrix operand order checked by reading, not run on hardware
//
#include <hip/hip_runtime.h>
#include <math.h>

#ifndef M_PI
#define M_PI 3.14159265358979323846
#endif

#define SS    128          // sheet units
#define P     (SS*SS)      // 16384 positions
#define AK    25
#define AC    2
#define DD    1250         // AC*AK*AK
#define LK    41
#define K2    1681         // LK*LK
#define CH64  82           // chunks: 41 kernel rows padded to 64 k each -> 82 x 32
#define MAXR  19
#define HALO  9            // MAXR/2
#define TW    34           // 16 + 2*HALO
#define ITERS 30
#define RET   152          // retina size S + AD*(AK-1)

typedef _Float16 f16;
typedef __attribute__((ext_vector_type(16))) _Float16 v16h;
typedef __attribute__((ext_vector_type(8)))  float    v8f;

// ---------------- static envelopes (computed on device, f64 like reference) --
__device__ inline double env_dist(int n, int i, int j) {
    double c = (n - 1) * 0.5;
    double a = i - c, b = j - c;
    return sqrt(a * a + b * b);
}

__global__ __launch_bounds__(1024) void k_env(float* SRE, float* MAXM, float* AFFE, float* LRI) {
    __shared__ float s_red[1024];
    __shared__ float s_lri[K2];
    int t = threadIdx.x;

    // SRE: rcos(9, 9)^2 * circle(9, 4.5), normalized to sum 1
    float sval = 0.f;
    if (t < 81) {
        int i = t / 9, j = t % 9;
        double d = env_dist(9, i, j);
        double r = cos(M_PI * d / 9.0);
        sval = (float)(r * r * (d < 4.5 ? 1.0 : 0.0));
    }
    s_red[t] = sval;
    __syncthreads();
    for (int s = 512; s > 0; s >>= 1) { if (t < s) s_red[t] += s_red[t + s]; __syncthreads(); }
    float ssum = s_red[0];
    __syncthreads();
    if (t < 81) SRE[t] = sval / ssum;

    // MAX_MASK: circle(19, 9.5)
    if (t < 361) {
        int i = t / 19, j = t % 19;
        MAXM[t] = env_dist(19, i, j) < 9.5 ? 1.f : 0.f;
    }
    // AFF_ENV: rcos(25,25)^2 * circle(25,12.5); max == 1 at center; tiled over 2 channels
    if (t < 625) {
        int i = t / 25, j = t % 25;
        double d = env_dist(25, i, j);
        double r = cos(M_PI * d / 25.0);
        float v = (float)(r * r * (d < 12.5 ? 1.0 : 0.0));
        AFFE[t] = v; AFFE[625 + t] = v;
    }
    // LRI_ENV: rcos(41,41)^2 * circle(41,20.5) * (1 - im), im = rcos(41,4.5)^2*circle(41,2.25)
    for (int k = t; k < K2; k += 1024) {
        int i = k / 41, j = k % 41;
        double d  = env_dist(41, i, j);
        double ri = cos(M_PI * d / 4.5);
        double im = ri * ri * (d < 2.25 ? 1.0 : 0.0);
        double rl = cos(M_PI * d / 41.0);
        s_lri[k] = (float)(rl * rl * (d < 20.5 ? 1.0 : 0.0) * (1.0 - im));
    }
    __syncthreads();
    float m = 0.f;
    for (int k = t; k < K2; k += 1024) m = fmaxf(m, s_lri[k]);
    s_red[t] = m;
    __syncthreads();
    for (int s = 512; s > 0; s >>= 1) { if (t < s) s_red[t] = fmaxf(s_red[t], s_red[t + s]); __syncthreads(); }
    float lmax = s_red[0];
    for (int k = t; k < K2; k += 1024) LRI[k] = s_lri[k] / lmax;
}

// ---------------- per-position weight row sums (normalizers fold to /rowsum) --
__global__ void k_rowsum(const float* __restrict__ rfs, const float* __restrict__ latw,
                         float* rsD, float* rsK) {
    int p = blockIdx.x * blockDim.x + threadIdx.x;
    if (p >= P) return;
    const float* r = rfs + (size_t)p * DD;
    float s = 0.f;
    for (int d = 0; d < DD; ++d) s += r[d];
    rsD[p] = s;
    const float* w = latw + (size_t)p * K2;
    float s2 = 0.f;
    for (int k = 0; k < K2; ++k) s2 += w[k];
    rsK[p] = s2;
}

// ---------------- pre-swizzle lateral weights into WMMA B-fragment layout -----
// Chunked over 64-padded kernel rows: chunk c covers ky=c>>1, kx = 32*(c&1)+0..31.
// B (32x16 f16): lane l holds column N=l&15; element e holds
// K = (l&16 ? 8:0) + (e<8 ? e : e+8) within the chunk (ISA 7.12.2).
// Entry = lat_weights[p, ky*41+kx] * LRI_ENV / rowsumK[p], or 0 for kx >= 41.
// The zeros annihilate the garbage A-entries read for padded kx.
__global__ void k_wswz(const float* __restrict__ latw, const float* __restrict__ lri,
                       const float* __restrict__ rsK, f16* __restrict__ wb) {
    int tid = blockIdx.x * blockDim.x + threadIdx.x;
    int lane = tid & 31;
    int idx = tid >> 5;                 // group*CH64 + chunk
    if (idx >= 1024 * CH64) return;
    int g = idx / CH64, c = idx - g * CH64;
    int n = lane & 15;
    int koff = (lane & 16) ? 8 : 0;
    int y = g >> 3, x0 = (g & 7) << 4;
    int p = y * SS + x0 + n;
    float inv = 1.0f / rsK[p];
    int ky = c >> 1;
    int kxb = ((c & 1) << 5) + koff;
    const float* wr = latw + (size_t)p * K2 + ky * 41;
    const float* lr = lri + ky * 41;
    f16 tmp[16] __attribute__((aligned(32)));
#pragma unroll
    for (int e = 0; e < 16; ++e) {
        int kx = kxb + (e < 8 ? e : e + 8);
        float v = (kx < 41) ? wr[kx] * lr[kx] * inv : 0.f;
        tmp[e] = (f16)v;
    }
    f16* dst = wb + ((size_t)idx * 32 + lane) * 16;
    const uint4* s4 = reinterpret_cast<const uint4*>(tmp);
    uint4* d4 = reinterpret_cast<uint4*>(dst);
    d4[0] = s4[0]; d4[1] = s4[1];
}

// ---------------- afferent response + state init ------------------------------
__global__ void k_aff(const float* __restrict__ x, const float* __restrict__ rfs,
                      const float* __restrict__ adath, const float* __restrict__ affe,
                      const float* __restrict__ rsD,
                      float* aff, float* lat, float* inh) {
    int p = blockIdx.x * blockDim.x + threadIdx.x;
    if (p >= P) return;
    int y = p >> 7, xx = p & 127;
    const float* rp = rfs + (size_t)p * DD;
    float s = 0.f;
    int d = 0;
    for (int c = 0; c < AC; ++c) {
        const float* xc = x + c * RET * RET;
        for (int ky = 0; ky < AK; ++ky) {
            const float* xr = xc + (y + ky) * RET + xx;
            for (int kx = 0; kx < AK; ++kx, ++d)
                s += xr[kx] * affe[d] * rp[d];
        }
    }
    float a = s / rsD[p] - adath[p];
    aff[p] = a;
    lat[p] = fmaxf(a, 0.f);
    inh[p] = 1.f;
}

// ---------------- 9x9 reflect-pad excitation conv -----------------------------
__global__ void k_le(const float* __restrict__ lat, const float* __restrict__ sre,
                     float* __restrict__ le) {
    __shared__ float s_sre[81];
    int t = threadIdx.x;
    if (t < 81) s_sre[t] = sre[t];
    __syncthreads();
    int p = blockIdx.x * blockDim.x + t;
    int y = p >> 7, x = p & 127;
    float acc = 0.f;
#pragma unroll
    for (int ky = 0; ky < 9; ++ky) {
        int yy = y + ky - 4;
        yy = yy < 0 ? -yy : (yy > 127 ? 254 - yy : yy);
#pragma unroll
        for (int kx = 0; kx < 9; ++kx) {
            int xx = x + kx - 4;
            xx = xx < 0 ? -xx : (xx > 127 ? 254 - xx : xx);
            acc += lat[yy * SS + xx] * s_sre[ky * 9 + kx];
        }
    }
    le[p] = acc;
}

// ---------------- lateral inhibition via diag-WMMA + pointwise new ------------
// Wave handles 16 positions of one row. A[i,k] = enveloped patch of position i,
// B[k,j] = weights of position j; neg_i = D[i,i].
// le is staged in LDS as f16, one 256-entry zero-padded row per virtual sheet
// row yb-40+vr (zero rows when out of sheet), so every A gather is an
// unconditional immediate-offset ds_load_u16.
// Processed 2 chunks per iteration (same LDS row, +64 half offset) with two
// independent accumulators / register sets to kill WMMA WAR hazard NOPs.
// Diagonal extracted branchlessly with 8 shuffles + cndmask selects.
template <bool PRE>
__global__ __launch_bounds__(512) void k_neg(const float* __restrict__ le,
                                             const f16* __restrict__ wb,
                                             const float* __restrict__ latw,  // fallback path
                                             const float* __restrict__ lri,
                                             const float* __restrict__ rsK,
                                             const float* __restrict__ aff,
                                             const float* __restrict__ inh,
                                             const float* __restrict__ lat,
                                             float* __restrict__ newb,
                                             float* __restrict__ diffb,
                                             int iter0) {
    __shared__ unsigned int s_buf[82 * 128];   // 82 rows x 256 f16 (data at halves [40,168))
    int t = threadIdx.x;
    int b = blockIdx.x;
    int yb = 2 * b;
    // zero the pad columns: dword cols [0,20) and [84,128) of every row
    for (int j = t; j < 82 * 64; j += 512) {
        int vr = j >> 6, q = j & 63;
        int col = q < 20 ? q : (64 + q);
        s_buf[vr * 128 + col] = 0u;
    }
    // data region: dword cols [20,84) <- f16(le[yb-40+vr, 2cx..2cx+1]) or zeros
    for (int j = t; j < 82 * 64; j += 512) {
        int vr = j >> 6, cx = j & 63;
        int yy = yb - 40 + vr;
        unsigned int pk = 0u;
        if ((unsigned)yy < 128u) {
            const float* rp = le + yy * SS + 2 * cx;
            union { f16 h[2]; unsigned int u; } cv;
            cv.h[0] = (f16)rp[0];
            cv.h[1] = (f16)rp[1];
            pk = cv.u;
        }
        s_buf[vr * 128 + 20 + cx] = pk;
    }
    __syncthreads();

    int w    = t >> 5;
    int lane = t & 31;
    int g    = b * 16 + w;
    int y    = g >> 3, x0 = (g & 7) << 4;
    int m    = lane & 15;
    int koff = (lane & 16) ? 8 : 0;
    const f16* sh = (const f16*)s_buf;
    int labase = x0 + m + 2 * koff;    // lane-constant half-index part
    int vrodd = y & 1;

    int pcol = y * SS + x0 + (lane & 15);        // fallback B column position
    float invk = PRE ? 0.f : (1.0f / rsK[pcol]);
    const float* wrow = latw + (size_t)pcol * K2;
    const f16* wgb = PRE ? (wb + (size_t)g * (CH64 * 512) + lane * 16) : (const f16*)0;

    v8f acc0 = {}, acc1 = {};
    for (int c = 0; c < CH64; c += 2) {
        // both chunks of the pair read the SAME LDS row (ky = c>>1)
        const f16* rowp = sh + ((vrodd + c) << 8) + labase;
        v16h a0, a1;
#pragma unroll
        for (int e = 0; e < 16; ++e) {
            int f = e < 8 ? e : e + 8;
            a0[e] = rowp[2 * f];
            a1[e] = rowp[64 + 2 * f];
        }
        v16h b0, b1;
        if (PRE) {
            b0 = *(const v16h*)(wgb + (size_t)c * 512);
            b1 = *(const v16h*)(wgb + (size_t)(c + 1) * 512);
            // speculative prefetch of the pair after next (c+4, c+5) (lane stride 64B spans 2048B)
            __builtin_prefetch(wgb - lane * 16 + (size_t)(c + 4) * 512 + lane * 32, 0, 0);
        } else {
            int ky = c >> 1;
            const float* wr = wrow + ky * 41;
            const float* lr = lri + ky * 41;
#pragma unroll
            for (int e = 0; e < 16; ++e) {
                int f = e < 8 ? e : e + 8;
                int kx0 = koff + f;
                int kx1 = 32 + koff + f;
                b0[e] = (f16)(wr[kx0] * lr[kx0] * invk);   // kx0 <= 31 always valid
                b1[e] = (f16)((kx1 < 41) ? wr[kx1] * lr[kx1] * invk : 0.f);
            }
        }
        acc0 = __builtin_amdgcn_wmma_f32_16x16x32_f16(
            false, a0, false, b0, (short)0, acc0, false, false);
        acc1 = __builtin_amdgcn_wmma_f32_16x16x32_f16(
            false, a1, false, b1, (short)0, acc1, false, false);
    }
    v8f acc = acc0 + acc1;
    // Branchless diagonal extraction:
    // D[i,i] = acc[i] @ lane i (i<8), acc[i-8] @ lane i+16 (i>=8).
    // Lane l (serving slot i=l&15) fetches each acc[r] from its source lane and
    // keeps the r it needs via cndmask selects.
    int src  = (lane < 8) ? lane : (lane + 16);
    int rsel = (lane < 8) ? lane : (lane - 8);   // only meaningful for lane<16
    float diag = 0.f;
#pragma unroll
    for (int r = 0; r < 8; ++r) {
        float v = __shfl(acc[r], src);
        diag = (rsel == r) ? v : diag;
    }
    if (lane < 16) {
        int p = y * SS + x0 + lane;
        float neg = diag;
        float lev = le[p];                       // exact f32 for the pointwise update
        float nv = fmaxf(lev * inh[p] - neg + aff[p], 0.f) * 6.0f;
        newb[p] = nv;
        float last = iter0 ? 0.f : lat[p];
        diffb[p] = fmaxf(nv - last, 0.f);
    }
}

// ---------------- 19x19 masked max-pools + inh update + lat <- new ------------
__global__ __launch_bounds__(256) void k_post(const float* __restrict__ newb,
                                              const float* __restrict__ diffb,
                                              const float* __restrict__ maxm,
                                              float* __restrict__ inh,
                                              float* __restrict__ lat) {
    __shared__ float s_new[TW * TW];
    __shared__ float s_dif[TW * TW];
    __shared__ float s_msk[361];
    int t = threadIdx.x;
    int bx = blockIdx.x & 7, by = blockIdx.x >> 3;
    int ox = bx * 16 - HALO, oy = by * 16 - HALO;
    for (int i = t; i < TW * TW; i += 256) {
        int iy = i / TW, ix = i - iy * TW;
        int yy = oy + iy, xx = ox + ix;
        bool inb = (unsigned)yy < 128u && (unsigned)xx < 128u;
        s_new[i] = inb ? newb[yy * SS + xx] : 0.f;
        s_dif[i] = inb ? diffb[yy * SS + xx] : 0.f;
    }
    for (int i = t; i < 361; i += 256) s_msk[i] = maxm[i];
    __syncthreads();
    int ty = t >> 4, tx = t & 15;
    float lmax = 0.f, smax = 0.f;
    for (int ky = 0; ky < MAXR; ++ky)
        for (int kx = 0; kx < MAXR; ++kx) {
            float msk = s_msk[ky * MAXR + kx];
            int i = (ty + ky) * TW + tx + kx;
            lmax = fmaxf(lmax, s_new[i] * msk);
            smax = fmaxf(smax, s_dif[i] * msk);
        }
    int p = (by * 16 + ty) * SS + bx * 16 + tx;
    float nv = s_new[(ty + HALO) * TW + tx + HALO];
    inh[p] = inh[p] * 0.5f + 0.5f * expf(-0.5f * lmax * lmax - 2.0f * smax * smax);
    lat[p] = nv;
}

extern "C" void kernel_launch(void* const* d_in, const int* in_sizes, int n_in,
                              void* d_out, int out_size, void* d_ws, size_t ws_size,
                              hipStream_t stream) {
    const float* x     = (const float*)d_in[0];   // [1,2,152,152]
    const float* rfs   = (const float*)d_in[1];   // [16384,1250,1]
    const float* latw  = (const float*)d_in[2];   // [16384,1681,1]
    const float* adath = (const float*)d_in[3];   // [1,1,128,128]
    float* out = (float*)d_out;

    float* wsf  = (float*)d_ws;
    float* SRE  = wsf + 0;        // 81
    float* MAXM = wsf + 128;      // 361
    float* AFFE = wsf + 512;      // 1250
    float* LRI  = wsf + 2048;     // 1681
    float* rsD  = wsf + 4096;
    float* rsK  = rsD + P;
    float* aff  = rsK + P;
    float* inh  = aff + P;
    float* lat  = inh + P;
    float* leb  = lat + P;
    float* newb = leb + P;
    float* difb = newb + P;
    f16*   wb   = (f16*)(difb + P);                            // byte off 540672, 32B aligned
    size_t need = 540672 + (size_t)1024 * CH64 * 512 * 2 + 8192;   // ~86.5 MB + prefetch slack
    bool pre = ws_size >= need;

    k_env<<<1, 1024, 0, stream>>>(SRE, MAXM, AFFE, LRI);
    k_rowsum<<<(P + 255) / 256, 256, 0, stream>>>(rfs, latw, rsD, rsK);
    if (pre)
        k_wswz<<<(1024 * CH64 * 32) / 256, 256, 0, stream>>>(latw, LRI, rsK, wb);
    k_aff<<<P / 256, 256, 0, stream>>>(x, rfs, adath, AFFE, rsD, aff, lat, inh);

    for (int it = 0; it < ITERS; ++it) {
        k_le<<<P / 256, 256, 0, stream>>>(lat, SRE, leb);
        if (pre)
            k_neg<true><<<64, 512, 0, stream>>>(leb, wb, latw, LRI, rsK,
                                                aff, inh, lat, newb, difb, it == 0 ? 1 : 0);
        else
            k_neg<false><<<64, 512, 0, stream>>>(leb, (const f16*)nullptr, latw, LRI, rsK,
                                                 aff, inh, lat, newb, difb, it == 0 ? 1 : 0);
        k_post<<<64, 256, 0, stream>>>(newb, difb, MAXM, inh, lat);
    }
    hipMemcpyAsync(out, lat, P * sizeof(float), hipMemcpyDeviceToDevice, stream);
}
